// BottomUpHTMM_71811853189751
// MI455X (gfx1250) — matrix operand order repaired and checked by reading, not compiled, not run on hardware
//
#include <hip/hip_runtime.h>

// ---------------- problem constants ----------------
#define KBR   8     // branching factor
#define CST   8     // hidden states
#define MSYM  32    // symbols
#define G     16    // generators
#define NT_NODES 299593   // nodes per tree = (8^7-1)/7
// level starts S(d) = (8^d-1)/7 : 0,1,9,73,585,4681,37449
#define TOTBLK 4682       // total ll-partial blocks across all level kernels
#define LN2F  0.69314718055994530942f

typedef __attribute__((ext_vector_type(16))) _Float16 v16h;
typedef __attribute__((ext_vector_type(8)))  float    v8f;

union FRAG { v16h h; _Float16 e[16]; uint4 q[2]; };

// ---------------------------------------------------------------------------
// Precompute: softmax reparameterizations.
//  Afrag : M^T pre-swizzled into WMMA A-fragment register order, f16.
//          value(i, K=l*8+j) = softmax_l(SP)[l,g] * softmax_i(A)[i,j,l,g]
//          lane (kh*16 + i), half (w*16 + h):  w=l>>2, kh=(l>>1)&1, h=j+((l&1)?8:0)
//          rows i=8..15 are zero padding.
//  sBsym[s][g][c] = softmax_s(B)[c,s,g]     (gather by symbol)
//  lPi  [g][l][c] = softmax_c(Pi)[c,l,g]    (leaf prior by position)
// ---------------------------------------------------------------------------
__global__ __launch_bounds__(256)
void htmm_pre(const float* __restrict__ A, const float* __restrict__ Bm,
              const float* __restrict__ Pi, const float* __restrict__ SP,
              _Float16* __restrict__ Afrag, float* __restrict__ sBsym,
              float* __restrict__ lPi)
{
    const int tid = threadIdx.x;
    for (int i = tid; i < G*32*32; i += 256) Afrag[i] = (_Float16)0.f;
    __syncthreads();

    // one thread per (j,l,g) column, softmax over parent state i
    for (int col = tid; col < 1024; col += 256) {
        int g = col & 15, l = (col >> 4) & 7, j = col >> 7;
        float mx = -1e30f;
        for (int i = 0; i < 8; ++i) mx = fmaxf(mx, A[((i*8 + j)*8 + l)*16 + g]);
        float e[8]; float den = 0.f;
        for (int i = 0; i < 8; ++i) { e[i] = __expf(A[((i*8 + j)*8 + l)*16 + g] - mx); den += e[i]; }
        float spmx = -1e30f;
        for (int q = 0; q < 8; ++q) spmx = fmaxf(spmx, SP[q*16 + g]);
        float spden = 0.f;
        for (int q = 0; q < 8; ++q) spden += __expf(SP[q*16 + g] - spmx);
        float s = (__expf(SP[l*16 + g] - spmx) / spden) / den;

        const int w  = l >> 2;
        const int kh = (l >> 1) & 1;
        const int h  = j + ((l & 1) ? 8 : 0);
        for (int i = 0; i < 8; ++i)
            Afrag[((size_t)(g*32 + kh*16 + i))*32 + w*16 + h] = (_Float16)(e[i] * s);
    }
    // smB: softmax over symbols per (c,g), stored [sym][g][c]
    for (int col = tid; col < 128; col += 256) {
        int g = col & 15, c = col >> 4;
        float mx = -1e30f;
        for (int s = 0; s < 32; ++s) mx = fmaxf(mx, Bm[(c*32 + s)*16 + g]);
        float den = 0.f;
        for (int s = 0; s < 32; ++s) den += __expf(Bm[(c*32 + s)*16 + g] - mx);
        float inv = 1.f / den;
        for (int s = 0; s < 32; ++s)
            sBsym[(s*16 + g)*8 + c] = __expf(Bm[(c*32 + s)*16 + g] - mx) * inv;
    }
    // smPi: softmax over states per (l,g), stored [g][l][c]
    for (int col = tid; col < 128; col += 256) {
        int g = col & 15, l = col >> 4;
        float mx = -1e30f;
        for (int c = 0; c < 8; ++c) mx = fmaxf(mx, Pi[(c*8 + l)*16 + g]);
        float den = 0.f;
        for (int c = 0; c < 8; ++c) den += __expf(Pi[(c*8 + l)*16 + g] - mx);
        float inv = 1.f / den;
        for (int c = 0; c < 8; ++c)
            lPi[(g*8 + l)*8 + c] = __expf(Pi[(c*8 + l)*16 + g] - mx) * inv;
    }
}

// ---------------------------------------------------------------------------
// One level transition, transposed mapping:  D[i, parent] = M^T x beta_children
//   A (v16h) : M^T fragment, preloaded in register order from Afrag
//   B (v16h) : child betas, K=64 x N=16 parents; lane n = parent column;
//              per K-chunk w lane needs children l = w*4+kh*2 and l+1
//              -> two contiguous b128 loads from f16 [node][g][c] buffer
//   D        : acc[0..7] = full 8-state vector of parent n, in ONE lane
//              (kh=0 lanes; kh=1 lanes hit the zero-padded rows)
// Post-WMMA normalization is in-lane: v_rcp_f32 for 1/nu, v_log_f32 (log2)
// accumulated and scaled by ln2 once in the final reduction.
// Block = 512 threads = 16 waves, wave g handles generator g, tile = 16 parents.
// ---------------------------------------------------------------------------
template<bool LEAF>
__global__ __launch_bounds__(512)
void htmm_level(const _Float16* __restrict__ Afrag,
                const float* __restrict__ gBsym,
                const float* __restrict__ gLPi,
                const _Float16* __restrict__ betaChild,
                _Float16* __restrict__ betaParent,
                const int* __restrict__ x,
                float* __restrict__ llPart,
                int U, int nPerTree, int childPerTree,
                int childStart, int parentStart)
{
    __shared__ float sB[MSYM*G*CST];   // 16 KB
    __shared__ float sPi[G*KBR*CST];   // 4 KB (leaf only)
    __shared__ int   psym[16];
    __shared__ int   csym[128];

    const int tid = threadIdx.x;
    const int tileBase = blockIdx.x * 16;

    for (int i = tid; i < MSYM*G*CST; i += 512) sB[i] = gBsym[i];
    if (LEAF) for (int i = tid; i < G*KBR*CST; i += 512) sPi[i] = gLPi[i];
    if (tid < 16) {
        int P = tileBase + tid; int sym = 0;
        if (P < U) {
            int tr = P >= nPerTree;
            sym = x[tr*NT_NODES + parentStart + P - tr*nPerTree];
        }
        psym[tid] = sym;
    }
    if (LEAF && tid < 128) {
        int row = tid >> 3, l = tid & 7;
        int P = tileBase + row; int sym = 0;
        if (P < U) {
            int cpos = P*8 + l;
            int tr = cpos >= childPerTree;
            sym = x[tr*NT_NODES + childStart + cpos - tr*childPerTree];
        }
        csym[tid] = sym;
    }
    __syncthreads();

    const int lane = tid & 31;
    const int g    = tid >> 5;
    const int n    = lane & 15;        // parent column this lane owns
    const int kh   = lane >> 4;
    const int P    = tileBase + n;
    float ll0 = 0.f, ll1 = 0.f;        // accumulated in log2 domain

    FRAG a[2]; FRAG b[2];
    #pragma unroll
    for (int w = 0; w < 2; ++w) {
        // A fragment: pre-swizzled M^T, contiguous per lane
        const uint4* ap = (const uint4*)(Afrag + ((size_t)(g*32 + lane))*32 + w*16);
        a[w].q[0] = ap[0];
        a[w].q[1] = ap[1];

        // B fragment: two consecutive children l0, l0+1 of parent n
        const int l0 = w*4 + kh*2;
        if (P < U) {
            if (LEAF) {
                #pragma unroll
                for (int half = 0; half < 2; ++half) {
                    const int l = l0 + half;
                    const int cpos = P*8 + l;
                    const int xs = csym[n*8 + l];
                    const float4* pi4 = (const float4*)&sPi[(g*8 + l)*8];
                    const float4* bb4 = (const float4*)&sB[(xs*16 + g)*8];
                    float4 p0 = pi4[0], p1 = pi4[1];
                    float4 q0 = bb4[0], q1 = bb4[1];
                    float pv[8];
                    pv[0] = p0.x*q0.x; pv[1] = p0.y*q0.y; pv[2] = p0.z*q0.z; pv[3] = p0.w*q0.w;
                    pv[4] = p1.x*q1.x; pv[5] = p1.y*q1.y; pv[6] = p1.z*q1.z; pv[7] = p1.w*q1.w;
                    const float nu = ((pv[0]+pv[1])+(pv[2]+pv[3])) + ((pv[4]+pv[5])+(pv[6]+pv[7]));
                    const float inv = __builtin_amdgcn_rcpf(nu);     // v_rcp_f32
                    #pragma unroll
                    for (int c = 0; c < 8; ++c)
                        b[w].e[half*8 + c] = (_Float16)(pv[c]*inv);
                    const float lg = __builtin_amdgcn_logf(nu);      // v_log_f32 (log2)
                    if (cpos >= childPerTree) ll1 += lg; else ll0 += lg;
                }
            } else {
                const size_t c0 = (size_t)(P*8 + l0);
                b[w].q[0] = *(const uint4*)(betaChild + (c0*G + g)*8);
                b[w].q[1] = *(const uint4*)(betaChild + ((c0 + 1)*G + g)*8);
            }
        } else {
            b[w].q[0] = uint4{0,0,0,0};
            b[w].q[1] = uint4{0,0,0,0};
        }
    }

    v8f acc = {0.f,0.f,0.f,0.f,0.f,0.f,0.f,0.f};
    acc = __builtin_amdgcn_wmma_f32_16x16x32_f16(false, a[0].h, false, b[0].h,
                                                 (short)0, acc, false, false);
    acc = __builtin_amdgcn_wmma_f32_16x16x32_f16(false, a[1].h, false, b[1].h,
                                                 (short)0, acc, false, false);

    // In-lane parent postprocessing: lane (kh=0, n) owns parent P entirely.
    if (kh == 0 && P < U) {
        const float4* br = (const float4*)&sB[(psym[n]*16 + g)*8];
        const float4 b0 = br[0], b1 = br[1];
        float bl[8];
        bl[0] = acc[0]*b0.x; bl[1] = acc[1]*b0.y; bl[2] = acc[2]*b0.z; bl[3] = acc[3]*b0.w;
        bl[4] = acc[4]*b1.x; bl[5] = acc[5]*b1.y; bl[6] = acc[6]*b1.z; bl[7] = acc[7]*b1.w;
        const float nu = ((bl[0]+bl[1])+(bl[2]+bl[3])) + ((bl[4]+bl[5])+(bl[6]+bl[7]));
        const float inv = __builtin_amdgcn_rcpf(nu);                 // v_rcp_f32
        union { _Float16 e[8]; uint4 q; } o;
        #pragma unroll
        for (int v = 0; v < 8; ++v) o.e[v] = (_Float16)(bl[v]*inv);
        *(uint4*)(betaParent + ((size_t)P*G + g)*8) = o.q;
        const float lg = __builtin_amdgcn_logf(nu);                  // log2
        if (P >= nPerTree) ll1 += lg; else ll0 += lg;
    }

    // deterministic in-wave xor butterfly via single-op ds_swizzle
    #define XADD(v, m) \
        v += __int_as_float(__builtin_amdgcn_ds_swizzle(__float_as_int(v), ((m) << 10) | 0x1f))
    XADD(ll0, 1);  XADD(ll1, 1);
    XADD(ll0, 2);  XADD(ll1, 2);
    XADD(ll0, 4);  XADD(ll1, 4);
    XADD(ll0, 8);  XADD(ll1, 8);
    XADD(ll0, 16); XADD(ll1, 16);
    #undef XADD

    if (lane == 0) {
        llPart[(size_t)blockIdx.x*32 + g*2 + 0] = ll0;
        llPart[(size_t)blockIdx.x*32 + g*2 + 1] = ll1;
    }
}

// Fixed-order final reduction; converts log2 sums to natural log.
__global__ __launch_bounds__(32)
void htmm_reduce(const float* __restrict__ llPart, float* __restrict__ out)
{
    const int t = threadIdx.x;          // t = g*2 + tree
    float acc = 0.f;
    for (int b = 0; b < TOTBLK; ++b) acc += llPart[(size_t)b*32 + t];
    out[(t & 1)*16 + (t >> 1)] = acc * LN2F;   // output layout [tree][g]
}

// ---------------------------------------------------------------------------
extern "C" void kernel_launch(void* const* d_in, const int* in_sizes, int n_in,
                              void* d_out, int out_size, void* d_ws, size_t ws_size,
                              hipStream_t stream) {
    (void)in_sizes; (void)n_in; (void)out_size; (void)ws_size;
    const float* A  = (const float*)d_in[0];
    const float* Bm = (const float*)d_in[1];
    const float* Pi = (const float*)d_in[2];
    const float* SP = (const float*)d_in[3];
    const int*   x  = (const int*)d_in[4];
    float* out = (float*)d_out;

    char* ws = (char*)d_ws;
    _Float16* Afrag = (_Float16*)(ws);                    //  32 KB
    float*    sBsym = (float*)(ws + 32768);               //  16 KB
    float*    lPi   = (float*)(ws + 49152);               //   4 KB
    float*    llP   = (float*)(ws + 53248);               // ~600 KB partials
    _Float16* buf0  = (_Float16*)(ws + (1u << 20));       //  16 MB ping
    _Float16* buf1  = (_Float16*)(ws + (1u << 20) + (17u << 20)); // 2 MB pong

    htmm_pre<<<1, 256, 0, stream>>>(A, Bm, Pi, SP, Afrag, sBsym, lPi);

    // transition d: children level d -> parents level d-1
    // args: U, parents/tree, children/tree(levelwide), S(d), S(d-1)
    htmm_level<true ><<<4096, 512, 0, stream>>>(Afrag, sBsym, lPi, nullptr, buf0, x,
                                                llP + (size_t)0*32,    65536, 32768, 262144, 37449, 4681);
    htmm_level<false><<< 512, 512, 0, stream>>>(Afrag, sBsym, lPi, buf0, buf1, x,
                                                llP + (size_t)4096*32,  8192,  4096,  32768,  4681,  585);
    htmm_level<false><<<  64, 512, 0, stream>>>(Afrag, sBsym, lPi, buf1, buf0, x,
                                                llP + (size_t)4608*32,  1024,   512,   4096,   585,   73);
    htmm_level<false><<<   8, 512, 0, stream>>>(Afrag, sBsym, lPi, buf0, buf1, x,
                                                llP + (size_t)4672*32,   128,    64,    512,    73,    9);
    htmm_level<false><<<   1, 512, 0, stream>>>(Afrag, sBsym, lPi, buf1, buf0, x,
                                                llP + (size_t)4680*32,    16,     8,     64,     9,    1);
    htmm_level<false><<<   1, 512, 0, stream>>>(Afrag, sBsym, lPi, buf0, buf1, x,
                                                llP + (size_t)4681*32,     2,     1,      8,     1,    0);

    htmm_reduce<<<1, 32, 0, stream>>>(llP, out);
}
